// PixelWiseAdpNet_7275674599979
// MI455X (gfx1250) — matrix operand order repaired and checked by reading, not compiled, not run on hardware
//
#include <hip/hip_runtime.h>

typedef __attribute__((ext_vector_type(8)))  float   v8f;
typedef __attribute__((ext_vector_type(8)))  __bf16  v8bf;
typedef __attribute__((ext_vector_type(16))) __bf16  v16bf;

#define NB 4
#define IN_CH 32
#define OUT_CH 64
#define FEAT_CH 256
#define P_TOT 32768          // S*OUT_H*OUT_W
#define NPIX 256             // B * AH * AW
#define PPATCH 512
#define CT 90688             // total params per pixel
// parameter layout offsets inside the per-pixel blob
#define W1_OFF 0             // 256 x 32
#define B1_OFF 8192          // 256
#define W2_OFF 8448          // 256 x 256
#define B2_OFF 73984         // 256
#define W3_OFF 74240         // 64 x 256
#define B3_OFF 90624         // 64

static __device__ __forceinline__ v8f wmma_bf16(v16bf a, v16bf b, v8f c) {
  return __builtin_amdgcn_wmma_f32_16x16x32_bf16(false, a, false, b, (short)0, c, false, false);
}

// Async global -> LDS copy of 16 bytes per lane (tracked by ASYNCcnt).
static __device__ __forceinline__ void async_copy_b128(unsigned lds_off, const void* gptr) {
  unsigned long long ga = (unsigned long long)(uintptr_t)gptr;
  asm volatile("global_load_async_to_lds_b128 %0, %1, off"
               :: "v"(lds_off), "v"(ga) : "memory");
}

static __device__ __forceinline__ void wait_asynccnt0() {
#if __has_builtin(__builtin_amdgcn_s_wait_asynccnt)
  __builtin_amdgcn_s_wait_asynccnt(0);
#else
  asm volatile("s_wait_asynccnt 0x0" ::: "memory");
#endif
}

// ---------------------------------------------------------------------------
// Kernel T: MLP_feature (B,256,8,8) -> Mt[pixel][ch] bf16   (pixel = b*64+hw)
// ---------------------------------------------------------------------------
__global__ __launch_bounds__(256) void k_transpose(const float* __restrict__ mlp,
                                                   __bf16* __restrict__ mt) {
  int pixel = blockIdx.x;
  int ch = threadIdx.x;
  int b = pixel >> 6, hw = pixel & 63;
  mt[pixel * 256 + ch] = (__bf16)mlp[((size_t)(b * 256 + ch)) * 64 + hw];
}

// ---------------------------------------------------------------------------
// Kernel 0: x0[pixel][plocal][32] = bf16( w_cd @ coord_data^T + b_cd + coord_em )
// WMMA: M = 32 (o), N = 16 points per wave-tile, K = 64
// ---------------------------------------------------------------------------
__global__ __launch_bounds__(256) void k_prep_x0(const float* __restrict__ coord_em,
                                                 const float* __restrict__ coord_data,
                                                 const float* __restrict__ w_cd,
                                                 const float* __restrict__ b_cd,
                                                 __bf16* __restrict__ x0) {
  const int lane  = threadIdx.x & 31;
  const int wave  = threadIdx.x >> 5;
  const int ptile = blockIdx.x * 8 + wave;       // 0..8191
  const int bp0   = ptile * 16;
  const int b     = bp0 >> 15;
  const int p0    = bp0 & (P_TOT - 1);
  const int row16 = lane & 15, sub = lane >> 4;

  const int p = p0 + row16;                      // this lane's point (N index)
  const float* cdrow = coord_data + ((size_t)b * P_TOT + p) * 64;

  v8f acc0 = {}; v8f acc1 = {};
  #pragma unroll
  for (int ks = 0; ks < 2; ++ks) {
    // B fragment: 16 contiguous k at fixed point
    v16bf bfr;
    const float* bp_ = cdrow + ks * 32 + sub * 16;
    #pragma unroll
    for (int j = 0; j < 16; ++j) bfr[j] = (__bf16)bp_[j];
    #pragma unroll
    for (int mt = 0; mt < 2; ++mt) {
      const float* ar = w_cd + (size_t)(mt * 16 + row16) * 64 + ks * 32 + sub * 16;
      v16bf afr;
      #pragma unroll
      for (int j = 0; j < 8; ++j) { afr[j] = (__bf16)ar[j]; afr[8 + j] = (__bf16)ar[8 + j]; }
      if (mt == 0) acc0 = wmma_bf16(afr, bfr, acc0);
      else         acc1 = wmma_bf16(afr, bfr, acc1);
    }
  }
  // epilogue: + b_cd + coord_em, permute (b,p) -> (pixel, plocal), store bf16
  int s = p >> 12, oh = (p >> 6) & 63, ow = p & 63;
  int pixel  = b * 64 + (oh >> 3) * 8 + (ow >> 3);
  int plocal = s * 64 + (oh & 7) * 8 + (ow & 7);
  const float* emb = coord_em + (size_t)b * IN_CH * P_TOT + p;
  __bf16* xrow = x0 + ((size_t)pixel * PPATCH + plocal) * IN_CH;
  #pragma unroll
  for (int r = 0; r < 8; ++r) {
    int m = r + ((lane >= 16) ? 8 : 0);
    { int o = m;       xrow[o] = (__bf16)(acc0[r] + b_cd[o] + emb[(size_t)o * P_TOT]); }
    { int o = 16 + m;  xrow[o] = (__bf16)(acc1[r] + b_cd[o] + emb[(size_t)o * P_TOT]); }
  }
}

// ---------------------------------------------------------------------------
// Kernel 1: feat[pixel][c] = bf16( w_feat @ Mt^T + b_feat )
// M = 90688 (c), N = 256 (pixels), K = 256. Block: 64-row A tile staged in LDS
// (with f32->bf16 conversion), 8 waves each own one 16-pixel N tile.
// ---------------------------------------------------------------------------
__global__ __launch_bounds__(256) void k_feat_gemm(const float* __restrict__ w_feat,
                                                   const float* __restrict__ b_feat,
                                                   const __bf16* __restrict__ mt,
                                                   __bf16* __restrict__ feat) {
  __shared__ __bf16 sA[64 * 256];                // 32 KB bf16 A tile
  const int tid  = threadIdx.x;
  const int lane = tid & 31, wave = tid >> 5;
  const int mgroup = blockIdx.x >> 1;            // 0..1416
  const int nthalf = blockIdx.x & 1;
  const int ntile  = nthalf * 8 + wave;          // 0..15
  const int c0     = mgroup * 64;
  const int row16  = lane & 15, sub = lane >> 4;

  // stage 64 x 256 fp32 -> bf16 (coalesced rows)
  for (int i = tid; i < 64 * 256; i += 256) {
    int r = i >> 8, k = i & 255;
    __builtin_prefetch(w_feat + (size_t)(c0 + 64 + r) * 256 + k, 0, 3);
    sA[i] = (__bf16)w_feat[(size_t)(c0 + r) * 256 + k];
  }
  __syncthreads();

  const int pixel = ntile * 16 + row16;
  const __bf16* brow = mt + (size_t)pixel * 256;

  v8f acc[4] = {v8f{}, v8f{}, v8f{}, v8f{}};
  #pragma unroll 2
  for (int ks = 0; ks < 8; ++ks) {
    v16bf bfr = *(const v16bf*)(brow + ks * 32 + sub * 16);
    #pragma unroll
    for (int m = 0; m < 4; ++m) {
      const __bf16* ar = sA + (size_t)(m * 16 + row16) * 256 + ks * 32 + sub * 16;
      v8bf lo = *(const v8bf*)(ar);
      v8bf hi = *(const v8bf*)(ar + 8);
      v16bf afr;
      #pragma unroll
      for (int j = 0; j < 8; ++j) { afr[j] = lo[j]; afr[8 + j] = hi[j]; }
      acc[m] = wmma_bf16(afr, bfr, acc[m]);
    }
  }
  __bf16* frow = feat + (size_t)pixel * CT;
  #pragma unroll
  for (int m = 0; m < 4; ++m) {
    #pragma unroll
    for (int r = 0; r < 8; ++r) {
      int c = c0 + m * 16 + r + ((lane >= 16) ? 8 : 0);
      frow[c] = (__bf16)(acc[m][r] + b_feat[c]);
    }
  }
}

// ---------------------------------------------------------------------------
// Kernel 2: per-pixel 3-layer MLP. One workgroup per pixel; full parameter
// blob (181 KB bf16) + x0 tile (32 KB) + 8 per-wave bounce tiles (64 KB) in LDS.
// Staging uses GLOBAL_LOAD_ASYNC_TO_LDS_B128 (ASYNCcnt) — no VGPR round-trip.
// Each wave carries 16-point row tiles through all layers.
// ---------------------------------------------------------------------------
__global__ __launch_bounds__(256) void k_pixel_mlp(const __bf16* __restrict__ feat,
                                                   const __bf16* __restrict__ x0,
                                                   float* __restrict__ out) {
  extern __shared__ char smem[];
  __bf16* blob = (__bf16*)smem;                        // 90688 bf16 = 181376 B
  __bf16* sX   = (__bf16*)(smem + 181376);             // 512*32  bf16 = 32768 B
  __bf16* sT   = (__bf16*)(smem + 181376 + 32768);     // 8*16*256 bf16 = 65536 B

  const int pixel = blockIdx.x;
  const int tid = threadIdx.x;
  const int lane = tid & 31, wave = tid >> 5;
  const int row16 = lane & 15, sub = lane >> 4;

  // async param + activation staging straight into LDS (16B per lane per op)
  {
    const unsigned blob_off = (unsigned)(uintptr_t)blob;   // flat LDS addr[31:0] = LDS offset
    const __bf16* src = feat + (size_t)pixel * CT;
    for (int i = tid; i < CT / 8; i += 256)
      async_copy_b128(blob_off + (unsigned)i * 16u, src + (size_t)i * 8);

    const unsigned sx_off = (unsigned)(uintptr_t)sX;
    const __bf16* sxp = x0 + (size_t)pixel * PPATCH * IN_CH;
    for (int i = tid; i < PPATCH * IN_CH / 8; i += 256)
      async_copy_b128(sx_off + (unsigned)i * 16u, sxp + (size_t)i * 8);
  }
  wait_asynccnt0();
  __syncthreads();

  const __bf16* sW1 = blob + W1_OFF;
  const __bf16* sb1 = blob + B1_OFF;
  const __bf16* sW2 = blob + W2_OFF;
  const __bf16* sb2 = blob + B2_OFF;
  const __bf16* sW3 = blob + W3_OFF;
  const __bf16* sb3 = blob + B3_OFF;

  __bf16* tile = sT + wave * 16 * 256;                 // this wave's [16][256] bounce
  const int b = pixel >> 6, hw = pixel & 63;
  const int ah = hw >> 3, aw = hw & 7;
  const int msel = (lane >= 16) ? 8 : 0;

  for (int mi = 0; mi < 4; ++mi) {
    const int mtile = mi * 8 + wave;                   // 0..31 row tile of 16 points

    // ---- layer 1: 32 -> 256, leaky relu ----
    v16bf a1;
    {
      const __bf16* xr = sX + (size_t)(mtile * 16 + row16) * IN_CH;
      v8bf lo = *(const v8bf*)(xr + sub * 16);
      v8bf hi = *(const v8bf*)(xr + 8 + sub * 16);
      #pragma unroll
      for (int j = 0; j < 8; ++j) { a1[j] = lo[j]; a1[8 + j] = hi[j]; }
    }
    #pragma unroll 4
    for (int nt = 0; nt < 16; ++nt) {
      v16bf bfr = *(const v16bf*)(sW1 + (size_t)(nt * 16 + row16) * 32 + sub * 16);
      v8f acc = {};
      acc = wmma_bf16(a1, bfr, acc);
      float bias = (float)sb1[nt * 16 + row16];
      #pragma unroll
      for (int r = 0; r < 8; ++r) {
        float v = acc[r] + bias;
        v = (v >= 0.f) ? v : 0.01f * v;
        tile[(r + msel) * 256 + nt * 16 + row16] = (__bf16)v;
      }
    }

    // ---- layer 2: 256 -> 256, leaky relu (A frags cached, tile overwritten) ----
    v16bf a2[8];
    #pragma unroll
    for (int ks = 0; ks < 8; ++ks) {
      const __bf16* xr = tile + row16 * 256 + ks * 32;
      v8bf lo = *(const v8bf*)(xr + sub * 16);
      v8bf hi = *(const v8bf*)(xr + 8 + sub * 16);
      #pragma unroll
      for (int j = 0; j < 8; ++j) { a2[ks][j] = lo[j]; a2[ks][8 + j] = hi[j]; }
    }
    #pragma unroll 2
    for (int nt = 0; nt < 16; ++nt) {
      v8f acc = {};
      #pragma unroll
      for (int ks = 0; ks < 8; ++ks) {
        v16bf bfr = *(const v16bf*)(sW2 + (size_t)(nt * 16 + row16) * 256 + ks * 32 + sub * 16);
        acc = wmma_bf16(a2[ks], bfr, acc);
      }
      float bias = (float)sb2[nt * 16 + row16];
      #pragma unroll
      for (int r = 0; r < 8; ++r) {
        float v = acc[r] + bias;
        v = (v >= 0.f) ? v : 0.01f * v;
        tile[(r + msel) * 256 + nt * 16 + row16] = (__bf16)v;
      }
    }

    // ---- layer 3: 256 -> 64, bias only, permuted store ----
    v16bf a3[8];
    #pragma unroll
    for (int ks = 0; ks < 8; ++ks) {
      const __bf16* xr = tile + row16 * 256 + ks * 32;
      v8bf lo = *(const v8bf*)(xr + sub * 16);
      v8bf hi = *(const v8bf*)(xr + 8 + sub * 16);
      #pragma unroll
      for (int j = 0; j < 8; ++j) { a3[ks][j] = lo[j]; a3[ks][8 + j] = hi[j]; }
    }
    #pragma unroll
    for (int nt = 0; nt < 4; ++nt) {
      v8f acc = {};
      #pragma unroll
      for (int ks = 0; ks < 8; ++ks) {
        v16bf bfr = *(const v16bf*)(sW3 + (size_t)(nt * 16 + row16) * 256 + ks * 32 + sub * 16);
        acc = wmma_bf16(a3[ks], bfr, acc);
      }
      int oc = nt * 16 + row16;
      float bias = (float)sb3[oc];
      #pragma unroll
      for (int r = 0; r < 8; ++r) {
        int pl = mtile * 16 + r + msel;                // point in patch
        int s = pl >> 6, ph = (pl >> 3) & 7, pw = pl & 7;
        size_t oidx = ((((size_t)(b * 64 + oc) * 8 + s) * 64) + ah * 8 + ph) * 64 + aw * 8 + pw;
        out[oidx] = acc[r] + bias;
      }
    }
  }
}

// ---------------------------------------------------------------------------
extern "C" void kernel_launch(void* const* d_in, const int* in_sizes, int n_in,
                              void* d_out, int out_size, void* d_ws, size_t ws_size,
                              hipStream_t stream) {
  const float* mlp = (const float*)d_in[0];
  const float* cem = (const float*)d_in[1];
  const float* cdt = (const float*)d_in[2];
  const float* wcd = (const float*)d_in[3];
  const float* bcd = (const float*)d_in[4];
  const float* wft = (const float*)d_in[5];
  const float* bft = (const float*)d_in[6];
  float* out = (float*)d_out;

  char* ws = (char*)d_ws;
  __bf16* mt   = (__bf16*)ws;                               // 131072 B
  __bf16* x0   = (__bf16*)(ws + 131072);                    // 8388608 B
  __bf16* feat = (__bf16*)(ws + 131072 + 8388608);          // 46432256 B

  k_transpose<<<NPIX, 256, 0, stream>>>(mlp, mt);
  k_prep_x0<<<1024, 256, 0, stream>>>(cem, cdt, wcd, bcd, x0);
  k_feat_gemm<<<2834, 256, 0, stream>>>(wft, bft, mt, feat);
  k_pixel_mlp<<<NPIX, 256, 279680, stream>>>(feat, x0, out);
}
